// MatrixPairwiseDistances_68959994905236
// MI455X (gfx1250) — matrix operand order repaired and checked by reading, not compiled, not run on hardware
//
#include <hip/hip_runtime.h>
#include <stdint.h>

typedef __attribute__((ext_vector_type(16))) __bf16 v16bf;
typedef __attribute__((ext_vector_type(8)))  float  v8f;
typedef int v4i __attribute__((vector_size(16)));

#define N_ROWS 8192
#define D_DIM  256
#define BTM    128   // block tile rows (x)
#define BTN    256   // block tile cols (y)
#define KT     32    // K per staged step (bf16)
#define LDST   40    // padded LDS row stride in bf16 elements (32 + 8)
#define LBUFA  (BTM * LDST)
#define LBUFB  (BTN * LDST)

#if defined(__gfx1250__) && __has_builtin(__builtin_amdgcn_global_load_async_to_lds_b128)
#define USE_ASYNC 1
#endif

union FragU { uint4 q[2]; v16bf v; };

static __device__ __forceinline__ unsigned short f2bf_rne(float f) {
    unsigned int u = __float_as_uint(f);
    u += 0x7FFFu + ((u >> 16) & 1u);   // RNE (inputs are finite normals)
    return (unsigned short)(u >> 16);
}
static __device__ __forceinline__ float bf2f(unsigned short h) {
    return __uint_as_float(((unsigned int)h) << 16);
}

static __device__ __forceinline__ v8f wmma_bf16(v16bf a, v16bf b, v8f c) {
    return __builtin_amdgcn_wmma_f32_16x16x32_bf16(false, a, false, b, (short)0, c, false, false);
}

// 16-byte global -> LDS copy (async on CDNA5 when available).
static __device__ __forceinline__ void cp16(unsigned short* l, const unsigned short* g) {
#ifdef USE_ASYNC
    __builtin_amdgcn_global_load_async_to_lds_b128(
        (__attribute__((address_space(1))) v4i*)(size_t)g,
        (__attribute__((address_space(3))) v4i*)(unsigned int)(size_t)l,
        0 /*offset*/, 0 /*cpol*/);
#else
    *(uint4*)l = *(const uint4*)g;
#endif
}

static __device__ __forceinline__ void wait_async0() {
#ifdef USE_ASYNC
#if __has_builtin(__builtin_amdgcn_s_wait_asynccnt)
    __builtin_amdgcn_s_wait_asynccnt(0);
#else
    asm volatile("s_wait_asynccnt 0" ::: "memory");
#endif
#endif
}

// ---------------------------------------------------------------------------
// Pre-pass: fused row norms + f32 -> bf16 hi/lo split (done ONCE per row).
// ---------------------------------------------------------------------------
__global__ __launch_bounds__(256) void pd_prep(const float* __restrict__ x,
                                               const float* __restrict__ y,
                                               float* __restrict__ xsq,
                                               float* __restrict__ ysq,
                                               unsigned short* __restrict__ Xhi,
                                               unsigned short* __restrict__ Xlo,
                                               unsigned short* __restrict__ Yhi,
                                               unsigned short* __restrict__ Ylo) {
    const int wave = threadIdx.x >> 5;
    const int lane = threadIdx.x & 31;
    int row = blockIdx.x * 8 + wave;               // 0 .. 2*N_ROWS-1
    const float* src; float* dst; unsigned short *hi, *lo;
    if (row < N_ROWS) { src = x; dst = xsq; hi = Xhi; lo = Xlo; }
    else { row -= N_ROWS; src = y; dst = ysq; hi = Yhi; lo = Ylo; }

    const float4* p = (const float4*)(src + (size_t)row * D_DIM);
    float4 a = p[lane * 2];
    float4 b = p[lane * 2 + 1];
    float f[8] = { a.x, a.y, a.z, a.w, b.x, b.y, b.z, b.w };

    unsigned short hs[8], ls[8];
    float s = 0.f;
    #pragma unroll
    for (int i = 0; i < 8; ++i) {
        hs[i] = f2bf_rne(f[i]);
        ls[i] = f2bf_rne(f[i] - bf2f(hs[i]));
        s += f[i] * f[i];
    }
    uint4 hq, lq;
    hq.x = (unsigned)hs[0] | ((unsigned)hs[1] << 16);
    hq.y = (unsigned)hs[2] | ((unsigned)hs[3] << 16);
    hq.z = (unsigned)hs[4] | ((unsigned)hs[5] << 16);
    hq.w = (unsigned)hs[6] | ((unsigned)hs[7] << 16);
    lq.x = (unsigned)ls[0] | ((unsigned)ls[1] << 16);
    lq.y = (unsigned)ls[2] | ((unsigned)ls[3] << 16);
    lq.z = (unsigned)ls[4] | ((unsigned)ls[5] << 16);
    lq.w = (unsigned)ls[6] | ((unsigned)ls[7] << 16);
    *(uint4*)(hi + (size_t)row * D_DIM + lane * 8) = hq;
    *(uint4*)(lo + (size_t)row * D_DIM + lane * 8) = lq;

    #pragma unroll
    for (int off = 16; off > 0; off >>= 1) s += __shfl_xor(s, off, 32);
    if (lane == 0) dst[row] = s;
}

// ---------------------------------------------------------------------------
// Main: dist = ||x||^2 + ||y||^2 - 2 x.yT via bf16x3-split WMMA GEMM.
// Block = 256 threads (8 waves, 2x4), 128x256 tile; wave sub-tile 64x64.
// Double-buffered LDS; async staging for step k+1 overlaps WMMA of step k.
// ---------------------------------------------------------------------------
__global__ __launch_bounds__(256) void pd_wmma(const unsigned short* __restrict__ Xhi,
                                               const unsigned short* __restrict__ Xlo,
                                               const unsigned short* __restrict__ Yhi,
                                               const unsigned short* __restrict__ Ylo,
                                               const float* __restrict__ xsq,
                                               const float* __restrict__ ysq,
                                               float* __restrict__ out) {
    __shared__ unsigned short AsHi[2 * LBUFA];
    __shared__ unsigned short AsLo[2 * LBUFA];
    __shared__ unsigned short BsHi[2 * LBUFB];
    __shared__ unsigned short BsLo[2 * LBUFB];

    const int tid  = threadIdx.x;
    const int wave = tid >> 5;
    const int lane = tid & 31;
    const int wrow = wave >> 2;          // 0..1  -> row offset wrow*64
    const int wcol = wave & 3;           // 0..3  -> col offset wcol*64
    const int half = lane >> 4;          // 0/1
    const int ln   = lane & 15;

    const int brow0 = blockIdx.y * BTM;
    const int bcol0 = blockIdx.x * BTN;

    v8f acc[4][4];
    #pragma unroll
    for (int mt = 0; mt < 4; ++mt)
        #pragma unroll
        for (int nt = 0; nt < 4; ++nt)
            #pragma unroll
            for (int r = 0; r < 8; ++r)
                acc[mt][nt][r] = 0.0f;

    // --- stage K-step `ks` into buffer `buf` -------------------------------
    // A tile: 128x32 bf16 = 512 16B-chunks (2/thread); B tile: 256x32 (4/thread)
    auto stage = [&](int ks, int buf) {
        const int kofs = ks * KT;
        #pragma unroll
        for (int i = 0; i < 2; ++i) {
            const int id  = tid + i * 256;
            const int row = id >> 2, c = id & 3;
            const int l   = buf * LBUFA + row * LDST + c * 8;
            const size_t g = (size_t)(brow0 + row) * D_DIM + kofs + c * 8;
            cp16(&AsHi[l], Xhi + g);
            cp16(&AsLo[l], Xlo + g);
        }
        #pragma unroll
        for (int i = 0; i < 4; ++i) {
            const int id  = tid + i * 256;
            const int row = id >> 2, c = id & 3;
            const int l   = buf * LBUFB + row * LDST + c * 8;
            const size_t g = (size_t)(bcol0 + row) * D_DIM + kofs + c * 8;
            cp16(&BsHi[l], Yhi + g);
            cp16(&BsLo[l], Ylo + g);
        }
    };

    // --- compute one K-step from buffer `buf` ------------------------------
    auto compute = [&](int buf) {
        const int ba = buf * LBUFA;
        const int bb = buf * LBUFB;
        FragU bh[4], bl[4];
        #pragma unroll
        for (int nt = 0; nt < 4; ++nt) {
            const int nrow = wcol * 64 + nt * 16 + ln;
            const int off  = bb + nrow * LDST + half * 16;
            bh[nt].q[0] = *(const uint4*)&BsHi[off];
            bh[nt].q[1] = *(const uint4*)&BsHi[off + 8];
            bl[nt].q[0] = *(const uint4*)&BsLo[off];
            bl[nt].q[1] = *(const uint4*)&BsLo[off + 8];
        }
        #pragma unroll
        for (int mt = 0; mt < 4; ++mt) {
            const int arow = wrow * 64 + mt * 16 + ln;
            const int o0 = ba + arow * LDST + half * 16;
            const int o1 = ba + arow * LDST + half * 16 + 8;
            FragU ah, al;
            ah.q[0] = *(const uint4*)&AsHi[o0];
            ah.q[1] = *(const uint4*)&AsHi[o1];
            al.q[0] = *(const uint4*)&AsLo[o0];
            al.q[1] = *(const uint4*)&AsLo[o1];
            #pragma unroll
            for (int nt = 0; nt < 4; ++nt) {
                acc[mt][nt] = wmma_bf16(ah.v, bh[nt].v, acc[mt][nt]);  // hi*hi
                acc[mt][nt] = wmma_bf16(ah.v, bl[nt].v, acc[mt][nt]);  // hi*lo
                acc[mt][nt] = wmma_bf16(al.v, bh[nt].v, acc[mt][nt]);  // lo*hi
            }
        }
    };

    // --- software pipeline -------------------------------------------------
    stage(0, 0);
    wait_async0();
    __syncthreads();

    #pragma unroll 1
    for (int ks = 0; ks < D_DIM / KT; ++ks) {
        if (ks + 1 < D_DIM / KT) stage(ks + 1, (ks + 1) & 1);  // prefetch next
        compute(ks & 1);
        wait_async0();      // this wave's async copies for ks+1 done
        __syncthreads();    // everyone's copies visible / reads drained
    }

    // --- epilogue: D layout VGPR r -> row (r + 8*half), col = ln -----------
    #pragma unroll
    for (int mt = 0; mt < 4; ++mt) {
        const int rbase = brow0 + wrow * 64 + mt * 16 + half * 8;
        float xs[8];
        #pragma unroll
        for (int r = 0; r < 8; ++r) xs[r] = xsq[rbase + r];
        #pragma unroll
        for (int nt = 0; nt < 4; ++nt) {
            const int col = bcol0 + wcol * 64 + nt * 16 + ln;
            const float ys = ysq[col];
            #pragma unroll
            for (int r = 0; r < 8; ++r) {
                out[(size_t)(rbase + r) * N_ROWS + col] =
                    xs[r] + ys - 2.0f * acc[mt][nt][r];
            }
        }
    }
}

extern "C" void kernel_launch(void* const* d_in, const int* in_sizes, int n_in,
                              void* d_out, int out_size, void* d_ws, size_t ws_size,
                              hipStream_t stream) {
    const float* x = (const float*)d_in[0];
    const float* y = (const float*)d_in[1];
    float* out = (float*)d_out;

    // Workspace layout: xsq(32KB) | ysq(32KB) | Xhi(4MB) | Xlo | Yhi | Ylo
    char* ws = (char*)d_ws;
    float* xsq = (float*)ws;
    float* ysq = (float*)(ws + 32 * 1024);
    unsigned short* Xhi = (unsigned short*)(ws + 64 * 1024);
    unsigned short* Xlo = Xhi + (size_t)N_ROWS * D_DIM;
    unsigned short* Yhi = Xlo + (size_t)N_ROWS * D_DIM;
    unsigned short* Ylo = Yhi + (size_t)N_ROWS * D_DIM;

    pd_prep<<<(2 * N_ROWS) / 8, 256, 0, stream>>>(x, y, xsq, ysq, Xhi, Xlo, Yhi, Ylo);

    dim3 grid(N_ROWS / BTN, N_ROWS / BTM);   // 32 x 64 blocks
    pd_wmma<<<grid, 256, 0, stream>>>(Xhi, Xlo, Yhi, Ylo, xsq, ysq, out);
}
